// Block_12068858101986
// MI455X (gfx1250) — matrix operand order repaired and checked by reading, not compiled, not run on hardware
//
#include <hip/hip_runtime.h>
#include <cstdint>

#define DIMC   1024
#define SEQ    2048
#define NTOK   4096      // B * N
#define NHEAD  16
#define HD     64
#define HIDDEN 4096

typedef __attribute__((ext_vector_type(16))) _Float16 v16h;
typedef __attribute__((ext_vector_type(8)))  float    v8f;
typedef __attribute__((ext_vector_type(4)))  uint32_t u32x4;

// ---------------------------------------------------------------------------
// Fragment loaders with contiguous k order (k = 16*half + s, same for A and B),
// vectorized: every fragment fills with two 128-bit loads.
//  A (16x32 f16): lane = half*16 + m ; VGPR v holds K pair half*16 + 2v
//    -> dwords v=0..7 contiguous at elem (row*lda + k0 + half*16)
//  B (32x16 f16): lane = group*16 + n ; VGPR v holds K pair group*16 + 2v
//  C/D (16x16 f32): lane L, VGPR v -> row = v + 8*(L/16), col = L%16
// ---------------------------------------------------------------------------

__device__ __forceinline__ void load_a_f16(v16h& a, const _Float16* A,
                                           int row0, int k0, int lda, int lane) {
  const int m = lane & 15, half = lane >> 4;
  const u32x4* A128 = reinterpret_cast<const u32x4*>(A);
  u32x4* a128 = reinterpret_cast<u32x4*>(&a);
  const int base = (row0 + m) * lda + k0 + half * 16;  // f16 element index, mult of 8
  a128[0] = A128[base >> 3];
  a128[1] = A128[(base >> 3) + 1];
}

// B from column-major dword-packed weights: Wp[n*(K/2) + (k>>1)] = {W[k][n], W[k+1][n]}
__device__ __forceinline__ void load_b_wp(v16h& b, const uint32_t* Wp,
                                          int k0, int n0, int Kh, int lane) {
  const int n = lane & 15, g = lane >> 4;
  const u32x4* P = reinterpret_cast<const u32x4*>(Wp);
  u32x4* b128 = reinterpret_cast<u32x4*>(&b);
  const int base = (n0 + n) * Kh + ((k0 + g * 16) >> 1);  // dword index, mult of 4
  b128[0] = P[base >> 2];
  b128[1] = P[(base >> 2) + 1];
}

// B where the WMMA K-dim is contiguous in memory (K-matrix for Q*K^T:
// B(k=dchan, n=token), memory row-major token x feature). 16 f16 contiguous.
__device__ __forceinline__ void load_b_kcontig(v16h& b, const _Float16* Km,
                                               int tok0, int d0, int ld, int lane) {
  const int n = lane & 15, g = lane >> 4;
  const u32x4* K128 = reinterpret_cast<const u32x4*>(Km);
  u32x4* b128 = reinterpret_cast<u32x4*>(&b);
  const int base = (tok0 + n) * ld + d0 + g * 16;  // f16 element index, mult of 8
  b128[0] = K128[base >> 3];
  b128[1] = K128[(base >> 3) + 1];
}

// B for P*V from token-pair-packed V: Vp[c*(SEQ/2) + (tok>>1)] = {V[tok][c], V[tok+1][c]}
__device__ __forceinline__ void load_b_vpack(v16h& b, const uint32_t* Vp,
                                             int tok0, int n0, int lane) {
  const int n = lane & 15, g = lane >> 4;
  const u32x4* P = reinterpret_cast<const u32x4*>(Vp);
  u32x4* b128 = reinterpret_cast<u32x4*>(&b);
  const int base = (n0 + n) * (SEQ / 2) + ((tok0 + g * 16) >> 1);  // dword idx
  b128[0] = P[base >> 2];
  b128[1] = P[(base >> 2) + 1];
}

// ---------------------------------------------------------------------------
// CDNA5 async global -> LDS copy (ASYNCcnt-tracked, CDNA5 ISA)
// ---------------------------------------------------------------------------
__device__ __forceinline__ void async_copy_b128(uint32_t lds_off, const void* gaddr) {
  asm volatile("global_load_async_to_lds_b128 %0, %1, off"
               :: "v"(lds_off), "v"(gaddr) : "memory");
}
__device__ __forceinline__ void wait_async0() {
  asm volatile("s_wait_asynccnt 0x0" ::: "memory");
}

// ---------------------------------------------------------------------------
// Weight pre-pack: fp32 (K x N) row-major -> f16 column-major dword-packed:
// Wp[n*(K/2) + kp] = {W[2kp][n], W[2kp+1][n]}
// ---------------------------------------------------------------------------
__global__ void pack_w(const float* __restrict__ W, uint32_t* __restrict__ Wp,
                       int K, int N) {
  const int idx = blockIdx.x * 256 + threadIdx.x;
  const int Kh = K >> 1;
  if (idx >= Kh * N) return;
  const int n = idx / Kh, kp = idx - n * Kh;
  union { _Float16 h[2]; uint32_t u; } cvt;
  cvt.h[0] = (_Float16)W[(2 * kp) * N + n];
  cvt.h[1] = (_Float16)W[(2 * kp + 1) * N + n];
  Wp[idx] = cvt.u;
}

// Repack the V third of QKV (f16, row-major token x 3072) into token-pair-
// packed layout per batch: Vp[b][c][tp] = {V[2tp][c], V[2tp+1][c]}, c in [0,1024)
__global__ void pack_v(const _Float16* __restrict__ qkv, uint32_t* __restrict__ Vp) {
  const int idx = blockIdx.x * 256 + threadIdx.x;  // total 2*1024*(SEQ/2)
  const int b = idx / (DIMC * (SEQ / 2));
  const int rem = idx - b * (DIMC * (SEQ / 2));
  const int c = rem / (SEQ / 2);
  const int tp = rem - c * (SEQ / 2);
  const _Float16* src = qkv + (size_t)b * SEQ * (3 * DIMC) + 2 * DIMC + c;
  union { _Float16 h[2]; uint32_t u; } cvt;
  cvt.h[0] = src[(2 * tp) * (3 * DIMC)];
  cvt.h[1] = src[(2 * tp + 1) * (3 * DIMC)];
  Vp[idx] = cvt.u;
}

// ---------------------------------------------------------------------------
// Fused LayerNorm (fp32 in) -> f16 out. One block per token row.
// ---------------------------------------------------------------------------
__global__ __launch_bounds__(256) void ln_f16(const float* __restrict__ x,
                                              const float* __restrict__ g,
                                              const float* __restrict__ bta,
                                              _Float16* __restrict__ out) {
  __shared__ float red[256];
  const int row = blockIdx.x;
  const float* xr = x + (size_t)row * DIMC;
  float s = 0.f, s2 = 0.f;
  for (int c = threadIdx.x; c < DIMC; c += 256) {
    const float v = xr[c];
    s += v; s2 += v * v;
  }
  red[threadIdx.x] = s; __syncthreads();
  for (int st = 128; st > 0; st >>= 1) {
    if ((int)threadIdx.x < st) red[threadIdx.x] += red[threadIdx.x + st];
    __syncthreads();
  }
  const float mean = red[0] * (1.0f / DIMC);
  __syncthreads();
  red[threadIdx.x] = s2; __syncthreads();
  for (int st = 128; st > 0; st >>= 1) {
    if ((int)threadIdx.x < st) red[threadIdx.x] += red[threadIdx.x + st];
    __syncthreads();
  }
  const float var = red[0] * (1.0f / DIMC) - mean * mean;
  const float inv = rsqrtf(var + 1e-5f);
  for (int c = threadIdx.x; c < DIMC; c += 256)
    out[(size_t)row * DIMC + c] = (_Float16)((xr[c] - mean) * inv * g[c] + bta[c]);
}

// ---------------------------------------------------------------------------
// Async-pipelined WMMA GEMM: C(MxN) = A(MxK f16 row-major) * Wpacked + bias.
// Block tile 128x128, K-step 32, double-buffered LDS tiles filled with
// global_load_async_to_lds_b128 (each tile loaded ONCE per block).
// 8 waves; wave tile 32x64 (8 WMMA / K-step, fragments via ds_load_b128).
// MODE 0: f16 out                       (QKV)
// MODE 1: f32 out = res + acc + bias    (proj, FC2)
// MODE 2: f16 out = gelu(acc + bias)    (FC1)
// ---------------------------------------------------------------------------
__device__ __forceinline__ void stage_tiles(const _Float16* A, const uint32_t* Wp,
                                            _Float16* AsBuf, uint32_t* BsBuf,
                                            int rowBlk, int colBlk, int k0,
                                            int K, int Kh, int tid) {
  // A tile: 128 rows x 32 f16 = 512 x 16B chunks
#pragma unroll
  for (int i = 0; i < 2; ++i) {
    const int idx = tid + i * 256;
    const int row = idx >> 2, quad = idx & 3;
    const _Float16* g = A + (rowBlk + row) * K + k0 + quad * 8;
    async_copy_b128((uint32_t)(uintptr_t)(AsBuf + row * 32 + quad * 8), g);
  }
  // B tile: 128 cols x 16 dwords = 512 x 16B chunks
#pragma unroll
  for (int i = 0; i < 2; ++i) {
    const int idx = tid + i * 256;
    const int n = idx >> 2, quad = idx & 3;
    const uint32_t* g = Wp + (colBlk + n) * Kh + (k0 >> 1) + quad * 4;
    async_copy_b128((uint32_t)(uintptr_t)(BsBuf + n * 16 + quad * 4), g);
  }
}

template <int MODE>
__global__ __launch_bounds__(256) void gemm_wmma(const _Float16* __restrict__ A,
                                                 const uint32_t* __restrict__ Wp,
                                                 const float* __restrict__ bias,
                                                 void* __restrict__ out,
                                                 const float* __restrict__ res,
                                                 int M, int Nn, int K) {
  __shared__ _Float16 As[2][128 * 32];
  __shared__ uint32_t Bs[2][128 * 16];
  const int tid  = threadIdx.x;
  const int lane = tid & 31;
  const int wid  = tid >> 5;
  const int rowBlk = blockIdx.y * 128;
  const int colBlk = blockIdx.x * 128;
  const int rw = (wid & 3) * 32;   // wave row offset in block tile
  const int cw = (wid >> 2) * 64;  // wave col offset in block tile
  const int Kh = K >> 1;
  (void)M;

  v8f acc[2][4] = {};
  stage_tiles(A, Wp, As[0], Bs[0], rowBlk, colBlk, 0, K, Kh, tid);

  int pb = 0;
  for (int k0 = 0; k0 < K; k0 += 32, pb ^= 1) {
    wait_async0();       // our async copies for buffer pb have landed
    __syncthreads();     // everyone's copies landed; prior reads of pb^1 drained
    if (k0 + 32 < K)
      stage_tiles(A, Wp, As[pb ^ 1], Bs[pb ^ 1], rowBlk, colBlk, k0 + 32, K, Kh, tid);

    v16h a0, a1;
    load_a_f16(a0, As[pb], rw,      0, 32, lane);
    load_a_f16(a1, As[pb], rw + 16, 0, 32, lane);
#pragma unroll
    for (int t = 0; t < 4; ++t) {
      v16h b;
      load_b_wp(b, Bs[pb], 0, cw + t * 16, 16, lane);
      acc[0][t] = __builtin_amdgcn_wmma_f32_16x16x32_f16(
          false, a0, false, b, (short)0, acc[0][t], false, false);
      acc[1][t] = __builtin_amdgcn_wmma_f32_16x16x32_f16(
          false, a1, false, b, (short)0, acc[1][t], false, false);
    }
  }

  const int n = lane & 15, half = lane >> 4;
  const int row0 = rowBlk + rw, col0 = colBlk + cw;
#pragma unroll
  for (int mm = 0; mm < 2; ++mm) {
#pragma unroll
    for (int t = 0; t < 4; ++t) {
      const int col = col0 + t * 16 + n;
      const float bv = bias[col];
#pragma unroll
      for (int v = 0; v < 8; ++v) {
        const int row = row0 + mm * 16 + v + 8 * half;
        const float val = acc[mm][t][v] + bv;
        if (MODE == 0) {
          ((_Float16*)out)[row * Nn + col] = (_Float16)val;
        } else if (MODE == 1) {
          ((float*)out)[row * Nn + col] = res[row * Nn + col] + val;
        } else {
          const float ge = 0.5f * val * (1.0f + erff(val * 0.70710678118f));
          ((_Float16*)out)[row * Nn + col] = (_Float16)ge;
        }
      }
    }
  }
}

// ---------------------------------------------------------------------------
// Flash attention: one wave handles a 16-row tile of one (batch, head).
// Per 32-token step: 4 WMMAs for S = Q*K^T, online softmax (shfl width-16
// row reductions), P routed C-layout -> A-layout through LDS, 4 WMMAs O+=P*V.
// All fragment loads are 128-bit.
// ---------------------------------------------------------------------------
__global__ __launch_bounds__(32) void attn_flash(const _Float16* __restrict__ qkv,
                                                 const uint32_t* __restrict__ Vpk,
                                                 _Float16* __restrict__ o) {
  __shared__ _Float16 pls[16 * 32];
  const int lane = threadIdx.x & 31;
  int tile = blockIdx.x;
  const int r0 = (tile & 127) * 16; tile >>= 7;
  const int h  = tile & 15;
  const int b  = tile >> 4;

  const _Float16* base = qkv + (size_t)b * SEQ * (3 * DIMC);
  const uint32_t* vbase = Vpk + (size_t)b * DIMC * (SEQ / 2);
  const int qc = h * HD;
  const int kc = DIMC + h * HD;
  const int vc = h * HD;            // column within packed-V slice
  const int ld = 3 * DIMC;

  v16h qa0, qa1;
  load_a_f16(qa0, base, r0, qc, ld, lane);
  load_a_f16(qa1, base, r0, qc + 32, ld, lane);

  v8f oacc[4] = {};
  float mrow[8], lrow[8];
#pragma unroll
  for (int v = 0; v < 8; ++v) { mrow[v] = -1e30f; lrow[v] = 0.f; }

  const float scale = 0.125f;  // 1/sqrt(64)
  const int n = lane & 15, half = lane >> 4;

  for (int j0 = 0; j0 < SEQ; j0 += 32) {
    v8f s0 = {}, s1 = {};
    {
      v16h bk;
      load_b_kcontig(bk, base, j0, kc, ld, lane);
      s0 = __builtin_amdgcn_wmma_f32_16x16x32_f16(false, qa0, false, bk, (short)0, s0, false, false);
      load_b_kcontig(bk, base, j0, kc + 32, ld, lane);
      s0 = __builtin_amdgcn_wmma_f32_16x16x32_f16(false, qa1, false, bk, (short)0, s0, false, false);
      load_b_kcontig(bk, base, j0 + 16, kc, ld, lane);
      s1 = __builtin_amdgcn_wmma_f32_16x16x32_f16(false, qa0, false, bk, (short)0, s1, false, false);
      load_b_kcontig(bk, base, j0 + 16, kc + 32, ld, lane);
      s1 = __builtin_amdgcn_wmma_f32_16x16x32_f16(false, qa1, false, bk, (short)0, s1, false, false);
    }

    float alpha[8];
#pragma unroll
    for (int v = 0; v < 8; ++v) {
      const float a0 = s0[v] * scale, a1 = s1[v] * scale;
      float mv = fmaxf(a0, a1);
#pragma unroll
      for (int off = 1; off < 16; off <<= 1)
        mv = fmaxf(mv, __shfl_xor(mv, off, 16));
      const float mn = fmaxf(mrow[v], mv);
      const float p0 = __expf(a0 - mn), p1 = __expf(a1 - mn);
      float rs = p0 + p1;
#pragma unroll
      for (int off = 1; off < 16; off <<= 1)
        rs += __shfl_xor(rs, off, 16);
      alpha[v] = __expf(mrow[v] - mn);
      lrow[v]  = lrow[v] * alpha[v] + rs;
      mrow[v]  = mn;
      const int r = v + 8 * half;
      pls[r * 32 + n]      = (_Float16)p0;
      pls[r * 32 + 16 + n] = (_Float16)p1;
    }
    __syncthreads();
    v16h pf;
    load_a_f16(pf, pls, 0, 0, 32, lane);  // 2x ds_load_b128
    __syncthreads();

#pragma unroll
    for (int t = 0; t < 4; ++t) {
#pragma unroll
      for (int v = 0; v < 8; ++v) oacc[t][v] *= alpha[v];
      v16h vb;
      load_b_vpack(vb, vbase, j0, vc + t * 16, lane);
      oacc[t] = __builtin_amdgcn_wmma_f32_16x16x32_f16(
          false, pf, false, vb, (short)0, oacc[t], false, false);
    }
  }

#pragma unroll
  for (int t = 0; t < 4; ++t) {
#pragma unroll
    for (int v = 0; v < 8; ++v) {
      const int r = r0 + v + 8 * half;
      o[((size_t)(b * SEQ + r)) * DIMC + h * HD + t * 16 + n] =
          (_Float16)(oacc[t][v] / lrow[v]);
    }
  }
}

// ---------------------------------------------------------------------------
// Host-side orchestration
// ---------------------------------------------------------------------------
extern "C" void kernel_launch(void* const* d_in, const int* in_sizes, int n_in,
                              void* d_out, int out_size, void* d_ws, size_t ws_size,
                              hipStream_t stream) {
  (void)in_sizes; (void)n_in; (void)out_size; (void)ws_size;

  const float* x      = (const float*)d_in[0];
  const float* ln1_g  = (const float*)d_in[1];
  const float* ln1_b  = (const float*)d_in[2];
  const float* qkv_w  = (const float*)d_in[3];
  const float* qkv_b  = (const float*)d_in[4];
  const float* proj_w = (const float*)d_in[5];
  const float* proj_b = (const float*)d_in[6];
  const float* ln2_g  = (const float*)d_in[7];
  const float* ln2_b  = (const float*)d_in[8];
  const float* fc1_w  = (const float*)d_in[9];
  const float* fc1_b  = (const float*)d_in[10];
  const float* fc2_w  = (const float*)d_in[11];
  const float* fc2_b  = (const float*)d_in[12];

  char* ws = (char*)d_ws;
  size_t off = 0;
  auto carve = [&](size_t bytes) -> char* {
    char* p = ws + off;
    off += (bytes + 255) & ~(size_t)255;
    return p;
  };

  uint32_t* qkvw_p = (uint32_t*)carve((size_t)(DIMC / 2) * (3 * DIMC) * 4);
  uint32_t* projw_p = (uint32_t*)carve((size_t)(DIMC / 2) * DIMC * 4);
  uint32_t* fc1w_p = (uint32_t*)carve((size_t)(DIMC / 2) * HIDDEN * 4);
  uint32_t* fc2w_p = (uint32_t*)carve((size_t)(HIDDEN / 2) * DIMC * 4);
  uint32_t* vpk    = (uint32_t*)carve((size_t)2 * DIMC * (SEQ / 2) * 4);
  _Float16* hA     = (_Float16*)carve((size_t)NTOK * DIMC * 2);
  _Float16* qkv_h  = (_Float16*)carve((size_t)NTOK * 3 * DIMC * 2);
  _Float16* attnO  = (_Float16*)carve((size_t)NTOK * DIMC * 2);
  float*    x2     = (float*)carve((size_t)NTOK * DIMC * 4);
  _Float16* h2     = (_Float16*)carve((size_t)NTOK * DIMC * 2);
  _Float16* ffn    = (_Float16*)carve((size_t)NTOK * HIDDEN * 2);

  // 1) pack weights (f32 row-major -> f16 col-major dword-paired)
  {
    int t;
    t = (DIMC / 2) * (3 * DIMC);
    pack_w<<<(t + 255) / 256, 256, 0, stream>>>(qkv_w, qkvw_p, DIMC, 3 * DIMC);
    t = (DIMC / 2) * DIMC;
    pack_w<<<(t + 255) / 256, 256, 0, stream>>>(proj_w, projw_p, DIMC, DIMC);
    t = (DIMC / 2) * HIDDEN;
    pack_w<<<(t + 255) / 256, 256, 0, stream>>>(fc1_w, fc1w_p, DIMC, HIDDEN);
    t = (HIDDEN / 2) * DIMC;
    pack_w<<<(t + 255) / 256, 256, 0, stream>>>(fc2_w, fc2w_p, HIDDEN, DIMC);
  }

  // 2) LN1 -> f16 activations
  ln_f16<<<NTOK, 256, 0, stream>>>(x, ln1_g, ln1_b, hA);

  // 3) QKV GEMM (4096 x 3072 x 1024), f16 out
  {
    dim3 grid((3 * DIMC) / 128, NTOK / 128);
    gemm_wmma<0><<<grid, 256, 0, stream>>>(hA, qkvw_p, qkv_b, (void*)qkv_h,
                                           nullptr, NTOK, 3 * DIMC, DIMC);
  }

  // 3b) repack V for 128-bit WMMA-B loads in attention
  {
    const int t = 2 * DIMC * (SEQ / 2);
    pack_v<<<(t + 255) / 256, 256, 0, stream>>>(qkv_h, vpk);
  }

  // 4) Flash attention (one wave per 16-row tile per head per batch)
  attn_flash<<<2 * NHEAD * (SEQ / 16), 32, 0, stream>>>(qkv_h, vpk, attnO);

  // 5) proj GEMM + residual(x) -> x2 (f32)
  {
    dim3 grid(DIMC / 128, NTOK / 128);
    gemm_wmma<1><<<grid, 256, 0, stream>>>(attnO, projw_p, proj_b, (void*)x2,
                                           x, NTOK, DIMC, DIMC);
  }

  // 6) LN2 -> f16
  ln_f16<<<NTOK, 256, 0, stream>>>(x2, ln2_g, ln2_b, h2);

  // 7) FC1 GEMM + exact GELU -> f16
  {
    dim3 grid(HIDDEN / 128, NTOK / 128);
    gemm_wmma<2><<<grid, 256, 0, stream>>>(h2, fc1w_p, fc1_b, (void*)ffn,
                                           nullptr, NTOK, HIDDEN, DIMC);
  }

  // 8) FC2 GEMM + residual(x2) -> d_out (f32)
  {
    dim3 grid(DIMC / 128, NTOK / 128);
    gemm_wmma<1><<<grid, 256, 0, stream>>>(ffn, fc2w_p, fc2_b, d_out,
                                           x2, NTOK, DIMC, HIDDEN);
  }
}
